// Gpt2Hyena_17798344474989
// MI455X (gfx1250) — matrix operand order repaired and checked by reading, not compiled, not run on hardware
//
#include <hip/hip_runtime.h>
#include <hip/hip_bf16.h>

// ---------------------------------------------------------------------------
// GPT-2 attention block for MI455X (gfx1250, wave32, WMMA bf16 16x16x32).
//   B=4, S=2048, D=1024, H=16, HD=64
// Software-pipelined GEMMs (32x64 wave tiles), flash attention with
// up-front fragment loads per column step.
// ---------------------------------------------------------------------------

typedef __attribute__((ext_vector_type(16))) __bf16        v16bf;
typedef __attribute__((ext_vector_type(8)))  float         v8f;
typedef __attribute__((ext_vector_type(4)))  unsigned int  v4u;
typedef __attribute__((ext_vector_type(4)))  float         v4f;

union FragBf { v4u q[2]; v16bf v; };

static __device__ __forceinline__ v8f wmma_bf16(FragBf a, FragBf b, v8f c) {
  // D = A(16x32 bf16) * B(32x16 bf16) + C(16x16 f32)
  return __builtin_amdgcn_wmma_f32_16x16x32_bf16(
      /*neg_a=*/false, a.v, /*neg_b=*/false, b.v,
      /*c_mod=*/(short)0, c, /*reuse_a=*/false, /*reuse_b=*/false);
}

#define S_LEN  2048
#define D_DIM  1024
#define NH     16
#define HD_DIM 64
#define BSZ    4
#define M_ROWS (BSZ * S_LEN)   // 8192
#define N_QKV  (3 * D_DIM)     // 3072

// ---------------------------------------------------------------------------
// Conversion / staging kernels
// ---------------------------------------------------------------------------
__global__ __launch_bounds__(256) void cvt_x_kernel(const float* __restrict__ x,
                                                    __bf16* __restrict__ xb) {
  int i = (blockIdx.x * 256 + threadIdx.x) * 4;
  v4f v = *(const v4f*)(x + i);
  xb[i + 0] = (__bf16)v.x;
  xb[i + 1] = (__bf16)v.y;
  xb[i + 2] = (__bf16)v.z;
  xb[i + 3] = (__bf16)v.w;
}

// w_attn (D, 3*H*HD) row-major f32 -> wT (3*H*HD, D) bf16  (B-transposed)
__global__ __launch_bounds__(256) void cvt_wattnT_kernel(const float* __restrict__ w,
                                                         __bf16* __restrict__ wT) {
  int o = blockIdx.x * 256 + threadIdx.x;      // o = n*1024 + d
  int n = o >> 10;
  int d = o & 1023;
  wT[o] = (__bf16)w[(size_t)d * N_QKV + n];
}

// w_proj (H*HD, D) row-major f32 -> wT (D, H*HD) bf16  (B-transposed)
__global__ __launch_bounds__(256) void cvt_wprojT_kernel(const float* __restrict__ w,
                                                         __bf16* __restrict__ wT) {
  int o = blockIdx.x * 256 + threadIdx.x;      // o = d*1024 + k
  int d = o >> 10;
  int k = o & 1023;
  wT[o] = (__bf16)w[(size_t)k * D_DIM + d];
}

// ---------------------------------------------------------------------------
// Pipelined GEMM core: one wave computes a 32(M) x 64(N) tile, K in steps
// of 32, with next-step fragments loaded before current-step WMMAs.
// ---------------------------------------------------------------------------
struct Tile32x64 { v8f acc[2][4]; };

static __device__ __forceinline__ void gemm_tile_32x64(
    const __bf16* __restrict__ A,   // [M][1024] row-major bf16
    const __bf16* __restrict__ BT,  // [N][1024] row-major bf16 (B transposed)
    int m0, int n0, int rl, int hi, Tile32x64& t) {
#pragma unroll
  for (int mb = 0; mb < 2; ++mb)
#pragma unroll
    for (int f = 0; f < 4; ++f)
#pragma unroll
      for (int r = 0; r < 8; ++r) t.acc[mb][f][r] = 0.0f;

  const __bf16* arow0 = A + (size_t)(m0 + rl) * D_DIM;
  const __bf16* arow1 = A + (size_t)(m0 + 16 + rl) * D_DIM;
  const __bf16* brow[4];
#pragma unroll
  for (int f = 0; f < 4; ++f)
    brow[f] = BT + (size_t)(n0 + f * 16 + rl) * D_DIM + 8 * hi;

  FragBf aC[2], bC[4];
  // prologue: k = 0
  aC[0].q[0] = *(const v4u*)(arow0 + 8 * hi);
  aC[0].q[1] = *(const v4u*)(arow0 + 16 + 8 * hi);
  aC[1].q[0] = *(const v4u*)(arow1 + 8 * hi);
  aC[1].q[1] = *(const v4u*)(arow1 + 16 + 8 * hi);
#pragma unroll
  for (int f = 0; f < 4; ++f) {
    bC[f].q[0] = ((const v4u*)brow[f])[0];
    bC[f].q[1] = ((const v4u*)brow[f])[2];
  }

  for (int k = 32; k < D_DIM; k += 32) {
    // issue next-step loads first so they overlap the WMMAs below
    FragBf aN[2], bN[4];
    aN[0].q[0] = *(const v4u*)(arow0 + k + 8 * hi);
    aN[0].q[1] = *(const v4u*)(arow0 + k + 16 + 8 * hi);
    aN[1].q[0] = *(const v4u*)(arow1 + k + 8 * hi);
    aN[1].q[1] = *(const v4u*)(arow1 + k + 16 + 8 * hi);
#pragma unroll
    for (int f = 0; f < 4; ++f) {
      bN[f].q[0] = ((const v4u*)(brow[f] + k))[0];
      bN[f].q[1] = ((const v4u*)(brow[f] + k))[2];
    }
    __builtin_prefetch(arow0 + k + 256, 0, 1);

#pragma unroll
    for (int f = 0; f < 4; ++f) {
      t.acc[0][f] = wmma_bf16(aC[0], bC[f], t.acc[0][f]);
      t.acc[1][f] = wmma_bf16(aC[1], bC[f], t.acc[1][f]);
    }
#pragma unroll
    for (int mb = 0; mb < 2; ++mb) aC[mb] = aN[mb];
#pragma unroll
    for (int f = 0; f < 4; ++f) bC[f] = bN[f];
  }
  // epilogue: last k-step
#pragma unroll
  for (int f = 0; f < 4; ++f) {
    t.acc[0][f] = wmma_bf16(aC[0], bC[f], t.acc[0][f]);
    t.acc[1][f] = wmma_bf16(aC[1], bC[f], t.acc[1][f]);
  }
}

// ---------------------------------------------------------------------------
// QKV GEMM:  (8192 x 1024) * (1024 x 3072) + b_attn
// Epilogue scatters q,k as [B,H,S,HD] bf16 and v transposed [B,H,HD,S] bf16.
// ---------------------------------------------------------------------------
__global__ __launch_bounds__(256) void qkv_gemm_kernel(
    const __bf16* __restrict__ A,    // x bf16 [8192][1024]
    const __bf16* __restrict__ BT,   // w_attn^T bf16 [3072][1024]
    const float*  __restrict__ bias, // b_attn f32 flat [3072]
    __bf16* __restrict__ qb, __bf16* __restrict__ kb, __bf16* __restrict__ vt) {
  const int lane = threadIdx.x & 31;
  const int w    = threadIdx.x >> 5;
  const int wg   = blockIdx.x * 8 + w;          // 12288 waves
  const int nt   = wg % 48;
  const int mt   = wg / 48;                     // < 256
  const int m0   = mt << 5;
  const int n0   = nt << 6;
  const int rl   = lane & 15;
  const int hi   = (lane >> 4) & 1;

  Tile32x64 t;
  gemm_tile_32x64(A, BT, m0, n0, rl, hi, t);

  const int b     = m0 >> 11;           // m0 / 2048
  const int sbase = m0 & 2047;
  const int qq    = n0 >> 10;           // 0=q, 1=k, 2=v
  const int h     = (n0 & 1023) >> 6;
  const size_t bh = (size_t)(b * NH + h);
#pragma unroll
  for (int mb = 0; mb < 2; ++mb) {
#pragma unroll
    for (int f = 0; f < 4; ++f) {
      const int e = f * 16 + rl;
      const float bval = bias[n0 + f * 16 + rl];
#pragma unroll
      for (int r = 0; r < 8; ++r) {
        const int s = sbase + mb * 16 + r + 8 * hi;
        const __bf16 val = (__bf16)(t.acc[mb][f][r] + bval);
        if (qq == 0)      qb[(bh * S_LEN + s) * HD_DIM + e] = val;
        else if (qq == 1) kb[(bh * S_LEN + s) * HD_DIM + e] = val;
        else              vt[(bh * HD_DIM + e) * S_LEN + s] = val;  // transposed
      }
    }
  }
}

// ---------------------------------------------------------------------------
// Flash attention (causal).  One wave owns 16 query rows; column tiles of 32.
// All k/v fragment loads issued up-front per step so the score WMMAs wait
// only on the k-group and the P*V WMMAs wait on nothing.
// ---------------------------------------------------------------------------
__global__ __launch_bounds__(256) void attn_kernel(
    const __bf16* __restrict__ qb,   // [B,H,S,HD]
    const __bf16* __restrict__ kb,   // [B,H,S,HD]
    const __bf16* __restrict__ vt,   // [B,H,HD,S]
    __bf16* __restrict__ ao) {       // [B,S,H*HD]
  __shared__ __align__(16) __bf16 pbuf[8][16 * 32];   // 1KB per wave

  const int lane = threadIdx.x & 31;
  const int w    = threadIdx.x >> 5;
  const int wg   = blockIdx.x * 8 + w;     // 8192 waves = 64 bh * 128 row tiles
  const int rowTile = wg & 127;
  const int bh      = wg >> 7;
  const int r0      = rowTile << 4;
  const int rl      = lane & 15;
  const int hi      = (lane >> 4) & 1;

  // q A-fragments (16 x 64 = two 16x32 fragments), loaded once
  const __bf16* qrow = qb + ((size_t)bh * S_LEN + r0 + rl) * HD_DIM;
  FragBf qa[2];
#pragma unroll
  for (int e = 0; e < 2; ++e) {
    const __bf16* bp = qrow + e * 32;
    qa[e].q[0] = *(const v4u*)(bp + 8 * hi);
    qa[e].q[1] = *(const v4u*)(bp + 16 + 8 * hi);
  }

  v8f O[4];
  float mst[8], lst[8];
#pragma unroll
  for (int f = 0; f < 4; ++f)
#pragma unroll
    for (int r = 0; r < 8; ++r) O[f][r] = 0.0f;
#pragma unroll
  for (int r = 0; r < 8; ++r) { mst[r] = -3.0e38f; lst[r] = 0.0f; }

  const __bf16* kbase = kb + (size_t)bh * S_LEN * HD_DIM;
  const __bf16* vbase = vt + (size_t)bh * HD_DIM * S_LEN;

  for (int c = 0; c < r0 + 16; c += 32) {
    // ---- issue ALL fragment loads for this step up-front ----
    FragBf kf[2][2];   // [ctile][e-step]
#pragma unroll
    for (int ct = 0; ct < 2; ++ct)
#pragma unroll
      for (int e = 0; e < 2; ++e) {
        const __bf16* bp = kbase + (size_t)(c + ct * 16 + rl) * HD_DIM + e * 32 + 8 * hi;
        kf[ct][e].q[0] = ((const v4u*)bp)[0];
        kf[ct][e].q[1] = ((const v4u*)bp)[2];
      }
    FragBf vf[4];
#pragma unroll
    for (int f = 0; f < 4; ++f) {
      const __bf16* bp = vbase + (size_t)(f * 16 + rl) * S_LEN + c + 8 * hi;
      vf[f].q[0] = ((const v4u*)bp)[0];
      vf[f].q[1] = ((const v4u*)bp)[2];
    }

    // ---- scores: 16 rows x 32 cols, over HD=64 ----
    v8f s[2];
#pragma unroll
    for (int ct = 0; ct < 2; ++ct) {
#pragma unroll
      for (int r = 0; r < 8; ++r) s[ct][r] = 0.0f;
#pragma unroll
      for (int e = 0; e < 2; ++e) s[ct] = wmma_bf16(qa[e], kf[ct][e], s[ct]);
    }

    // ---- scale + causal mask + row max (xor-reduce across 16-lane half) ----
    float rowm[8];
#pragma unroll
    for (int r = 0; r < 8; ++r) {
      const int row  = r0 + r + 8 * hi;
      const int col0 = c + rl;
      const int col1 = c + 16 + rl;
      float a0 = s[0][r] * 0.125f;
      float a1 = s[1][r] * 0.125f;
      if (col0 > row) a0 = -1.0e9f;
      if (col1 > row) a1 = -1.0e9f;
      s[0][r] = a0; s[1][r] = a1;
      float m = fmaxf(a0, a1);
      m = fmaxf(m, __shfl_xor(m, 1, 32));
      m = fmaxf(m, __shfl_xor(m, 2, 32));
      m = fmaxf(m, __shfl_xor(m, 4, 32));
      m = fmaxf(m, __shfl_xor(m, 8, 32));
      rowm[r] = m;
    }

    // ---- online softmax update, P -> LDS (C-layout -> row-major tile) ----
#pragma unroll
    for (int r = 0; r < 8; ++r) {
      const float mnew = fmaxf(mst[r], rowm[r]);
      const float corr = __expf(mst[r] - mnew);
      const float p0   = __expf(s[0][r] - mnew);
      const float p1   = __expf(s[1][r] - mnew);
      mst[r] = mnew;
      float rs = p0 + p1;
      rs += __shfl_xor(rs, 1, 32);
      rs += __shfl_xor(rs, 2, 32);
      rs += __shfl_xor(rs, 4, 32);
      rs += __shfl_xor(rs, 8, 32);
      lst[r] = lst[r] * corr + rs;
#pragma unroll
      for (int f = 0; f < 4; ++f) O[f][r] *= corr;
      const int rowl = r + 8 * hi;
      pbuf[w][rowl * 32 + rl]      = (__bf16)p0;
      pbuf[w][rowl * 32 + 16 + rl] = (__bf16)p1;
    }
    // Same-wave LDS RAW: DS ops are in-order; fence stops compiler reordering.
    asm volatile("" ::: "memory");

    // ---- reload P as A fragment (16x32) ----
    FragBf pa;
    {
      const __bf16* lp = &pbuf[w][rl * 32];
      pa.q[0] = *(const v4u*)(lp + 8 * hi);
      pa.q[1] = *(const v4u*)(lp + 16 + 8 * hi);
    }

    // ---- O += P * V  (4 N-tiles of 16 over HD=64); vf already resident ----
#pragma unroll
    for (int f = 0; f < 4; ++f) O[f] = wmma_bf16(pa, vf[f], O[f]);
  }

  // ---- finalize: divide by l, store [B,S,H*HD] bf16 ----
  const int b = bh >> 4;
  const int h = bh & 15;
#pragma unroll
  for (int r = 0; r < 8; ++r) {
    const float inv = 1.0f / lst[r];
    const int   srow = r0 + r + 8 * hi;
    const size_t obase = ((size_t)(b * S_LEN + srow)) * D_DIM + h * HD_DIM;
#pragma unroll
    for (int f = 0; f < 4; ++f)
      ao[obase + f * 16 + rl] = (__bf16)(O[f][r] * inv);
  }
}

// ---------------------------------------------------------------------------
// Output projection: (8192 x 1024) * (1024 x 1024) + b_proj -> f32 d_out
// ---------------------------------------------------------------------------
__global__ __launch_bounds__(256) void proj_gemm_kernel(
    const __bf16* __restrict__ A,    // attn out bf16 [8192][1024]
    const __bf16* __restrict__ BT,   // w_proj^T bf16 [1024][1024]
    const float*  __restrict__ bias, // b_proj [1024]
    float* __restrict__ out) {       // [8192][1024] f32
  const int lane = threadIdx.x & 31;
  const int w    = threadIdx.x >> 5;
  const int wg   = blockIdx.x * 8 + w;   // 4096 waves
  const int nt   = wg & 15;
  const int mt   = wg >> 4;              // < 256
  const int m0   = mt << 5;
  const int n0   = nt << 6;
  const int rl   = lane & 15;
  const int hi   = (lane >> 4) & 1;

  Tile32x64 t;
  gemm_tile_32x64(A, BT, m0, n0, rl, hi, t);

#pragma unroll
  for (int mb = 0; mb < 2; ++mb) {
#pragma unroll
    for (int f = 0; f < 4; ++f) {
      const int n = n0 + f * 16 + rl;
      const float bval = bias[n];
#pragma unroll
      for (int r = 0; r < 8; ++r) {
        const int m = m0 + mb * 16 + r + 8 * hi;
        out[(size_t)m * D_DIM + n] = t.acc[mb][f][r] + bval;
      }
    }
  }
}

// ---------------------------------------------------------------------------
// Launch
// ---------------------------------------------------------------------------
extern "C" void kernel_launch(void* const* d_in, const int* in_sizes, int n_in,
                              void* d_out, int out_size, void* d_ws, size_t ws_size,
                              hipStream_t stream) {
  (void)in_sizes; (void)n_in; (void)out_size; (void)ws_size;
  const float* x      = (const float*)d_in[0];   // (4,2048,1024)
  const float* w_attn = (const float*)d_in[1];   // (1024,3,16,64)
  const float* b_attn = (const float*)d_in[2];   // (3,16,64)
  const float* w_proj = (const float*)d_in[3];   // (16,64,1024)
  const float* b_proj = (const float*)d_in[4];   // (1024,)
  float* out = (float*)d_out;

  char* ws = (char*)d_ws;
  constexpr size_t SZ_XB  = (size_t)M_ROWS * D_DIM * 2;            // 16.78 MB
  constexpr size_t SZ_WAT = (size_t)N_QKV * D_DIM * 2;             //  6.29 MB
  constexpr size_t SZ_WPT = (size_t)D_DIM * D_DIM * 2;             //  2.10 MB
  constexpr size_t SZ_QKV = (size_t)BSZ * NH * S_LEN * HD_DIM * 2; // 16.78 MB each
  __bf16* xb  = (__bf16*)(ws);
  __bf16* waT = (__bf16*)(ws + SZ_XB);
  __bf16* wpT = (__bf16*)(ws + SZ_XB + SZ_WAT);
  __bf16* qb  = (__bf16*)(ws + SZ_XB + SZ_WAT + SZ_WPT);
  __bf16* kb  = (__bf16*)(ws + SZ_XB + SZ_WAT + SZ_WPT + SZ_QKV);
  __bf16* vt  = (__bf16*)(ws + SZ_XB + SZ_WAT + SZ_WPT + 2 * SZ_QKV);
  __bf16* ao  = (__bf16*)(ws + SZ_XB + SZ_WAT + SZ_WPT + 3 * SZ_QKV);

  // Stage inputs to bf16 (weights transposed for contiguous B-fragment loads)
  cvt_x_kernel<<<(M_ROWS * D_DIM) / 1024, 256, 0, stream>>>(x, xb);
  cvt_wattnT_kernel<<<(N_QKV * D_DIM) / 256, 256, 0, stream>>>(w_attn, waT);
  cvt_wprojT_kernel<<<(D_DIM * D_DIM) / 256, 256, 0, stream>>>(w_proj, wpT);

  // QKV projection: 256 M32-tiles * 48 N64-tiles = 12288 waves / 8 = 1536 blocks
  qkv_gemm_kernel<<<1536, 256, 0, stream>>>(xb, waT, b_attn, qb, kb, vt);

  // Flash attention: 64 (b,h) * 128 row-tiles = 8192 waves / 8 = 1024 blocks
  attn_kernel<<<1024, 256, 0, stream>>>(qb, kb, vt, ao);

  // Output projection: 256 M32-tiles * 16 N64-tiles = 4096 waves / 8 = 512 blocks
  proj_gemm_kernel<<<512, 256, 0, stream>>>(ao, wpT, b_proj, out);
}
